// LearnedChannelSelfAttentionPooling_43052752175582
// MI455X (gfx1250) — matrix operand order repaired and checked by reading, compiled, not run on hardware
//
#include <hip/hip_runtime.h>

// Problem constants (from reference)
#define Bc   2
#define Cc   72
#define Nc   (96 * 96)     // 9216 spatial positions
#define DP   96            // C padded to 3 * 32 (bf16 WMMA K-chunks)
#define COo  18            // C / FACTOR
#define FACT 4

typedef __attribute__((ext_vector_type(8)))  float  v8f;
typedef __attribute__((ext_vector_type(16))) __bf16 v16bf;

union V16 { v16bf v; uint4 u[2]; unsigned w[8]; };

__device__ __forceinline__ unsigned short f2bf(float f) {
  union { float f; unsigned u; } cv; cv.f = f;
  unsigned r = cv.u + 0x7FFFu + ((cv.u >> 16) & 1u);   // round-to-nearest-even
  return (unsigned short)(r >> 16);
}

// pack two floats into one dword of bf16
__device__ __forceinline__ unsigned packbf(float a, float b) {
#if __has_builtin(__builtin_amdgcn_cvt_pk_bf16_f32)
  typedef __attribute__((ext_vector_type(2))) __bf16 v2bf;
  union { v2bf v; unsigned u; } cv;
  cv.v = __builtin_amdgcn_cvt_pk_bf16_f32(a, b);
  return cv.u;
#else
  union { float f; unsigned u; } x, y; x.f = a; y.f = b;
  return ((x.u + 0x8000u) >> 16) | ((y.u + 0x8000u) & 0xFFFF0000u);
#endif
}

// -------------------------------------------------------------------------
// Stage 1: q/k/v = W * x + b, emitted in WMMA-friendly bf16 layouts.
//   qws, kws : [B][N][DP]  row-major per pixel, channels 72..95 zero-padded
//   vws      : [B][DP][N]  channel-major; channel 72 is set to 1.0 so the
//              PV WMMA accumulates the softmax denominator for free.
// -------------------------------------------------------------------------
__global__ __launch_bounds__(256) void qkv_kernel(
    const float* __restrict__ x,
    const float* __restrict__ Wq, const float* __restrict__ bq,
    const float* __restrict__ Wk, const float* __restrict__ bk,
    const float* __restrict__ Wv, const float* __restrict__ bv,
    unsigned short* __restrict__ qws,
    unsigned short* __restrict__ kws,
    unsigned short* __restrict__ vws)
{
  size_t id = (size_t)blockIdx.x * 256 + threadIdx.x;   // (b*DP + d)*N + n
  int n = (int)(id % Nc);
  size_t t = id / Nc;
  int d = (int)(t % DP);
  int b = (int)(t / DP);

  float aq = 0.f, ak = 0.f, av = 0.f;
  if (d < Cc) {
    const float* xc = x + (size_t)b * Cc * Nc + n;
    const float* wqr = Wq + (size_t)d * Cc;
    const float* wkr = Wk + (size_t)d * Cc;
    const float* wvr = Wv + (size_t)d * Cc;
#pragma unroll 8
    for (int c = 0; c < Cc; ++c) {
      float xv = xc[(size_t)c * Nc];    // coalesced across n; W reads are scalar
      aq = fmaf(wqr[c], xv, aq);
      ak = fmaf(wkr[c], xv, ak);
      av = fmaf(wvr[c], xv, av);
    }
    aq += bq[d]; ak += bk[d]; av += bv[d];
  }
  qws[((size_t)b * Nc + n) * DP + d] = f2bf(aq);
  kws[((size_t)b * Nc + n) * DP + d] = f2bf(ak);
  vws[((size_t)b * DP + d) * Nc + n] = (d == Cc) ? f2bf(1.0f) : f2bf(av);
}

// -------------------------------------------------------------------------
// Stage 2: fused flash attention, fully transposed tiles (LDS-free).
//   S^T = K_tile x Q^T   (two 16x16 tiles per 32-key step, 3 K-chunks)
//   O^T += V^T x P^T     (6 channel chunks; channel 72 == ones -> denominator)
// 4 waves / block, 16 queries / wave, bf16 WMMA with f32 accumulation.
// -------------------------------------------------------------------------
__global__ __launch_bounds__(128) void attn_kernel(
    const unsigned short* __restrict__ qws,
    const unsigned short* __restrict__ kws,
    const unsigned short* __restrict__ vws,
    float* __restrict__ ows)
{
  const int wave    = threadIdx.x >> 5;
  const int lane    = threadIdx.x & 31;
  const int h       = lane >> 4;        // 16-lane half
  const int lcol    = lane & 15;        // this lane's query column
  const int blkPerB = Nc / 64;
  const int b       = blockIdx.x / blkPerB;
  const int qbase   = (blockIdx.x % blkPerB) * 64 + wave * 16;

  // --- Q^T as B-operands (contiguous k order, same as A): lane col = query lcol,
  //     K = chunk*32 + h*16 + {0..15} contiguous.
  V16 bq[3];
  {
    const unsigned short* qcol =
        qws + ((size_t)b * Nc + qbase + lcol) * DP + 16 * h;
#pragma unroll
    for (int c = 0; c < 3; ++c) {
      bq[c].u[0] = *(const uint4*)(qcol + c * 32);
      bq[c].u[1] = *(const uint4*)(qcol + c * 32 + 8);
    }
  }

  v8f oacc[6];
#pragma unroll
  for (int d = 0; d < 6; ++d) oacc[d] = v8f{};
  float m = -3.0e30f;                   // running max, per query (per lane)

  const int koff = 16 * h;              // A-layout K split: {0..15} / {16..31} contiguous

  for (int kb = 0; kb < Nc; kb += 32) {
    // --- S^T tiles: A = K rows (lane row = key), B = Q^T
    const unsigned short* krow0 = kws + ((size_t)b * Nc + kb + lcol) * DP;
    const unsigned short* krow1 = krow0 + 16 * DP;
    __builtin_prefetch(krow0 + 32 * DP, 0, 1);         // next key tile
    v8f s0 = v8f{}, s1 = v8f{};
#pragma unroll
    for (int c = 0; c < 3; ++c) {
      V16 ak0, ak1;
      ak0.u[0] = *(const uint4*)(krow0 + c * 32 + koff);
      ak0.u[1] = *(const uint4*)(krow0 + c * 32 + 8 + koff);
      ak1.u[0] = *(const uint4*)(krow1 + c * 32 + koff);
      ak1.u[1] = *(const uint4*)(krow1 + c * 32 + 8 + koff);
      s0 = __builtin_amdgcn_wmma_f32_16x16x32_bf16(false, ak0.v, false, bq[c].v,
                                                   (short)0, s0, false, false);
      s1 = __builtin_amdgcn_wmma_f32_16x16x32_bf16(false, ak1.v, false, bq[c].v,
                                                   (short)0, s1, false, false);
    }

    // --- online softmax over keys (lane-local + one xor-16 combine).
    float mt = fmaxf(s0[0], s1[0]);
#pragma unroll
    for (int r = 1; r < 8; ++r) mt = fmaxf(mt, fmaxf(s0[r], s1[r]));
    mt = fmaxf(mt, __shfl_xor(mt, 16, 32));
    if (__any(mt > m)) {                // rare once running max settles
      float mn = fmaxf(m, mt);
      float sc = __expf(m - mn);        // == 1 for lanes that didn't grow
      m = mn;
#pragma unroll
      for (int d = 0; d < 6; ++d)
#pragma unroll
        for (int r = 0; r < 8; ++r) oacc[d][r] *= sc;
    }
#pragma unroll
    for (int r = 0; r < 8; ++r) {
      s0[r] = __expf(s0[r] - m);
      s1[r] = __expf(s1[r] - m);
    }

    // --- P^T into B-operand layout. Pack to bf16 dwords first (order is
    //     already key-ascending per lane), then swap 4 dwords with the
    //     xor-16 partner: lane needs keys {16h..16h+15} for its query.
    unsigned p0[4], p1[4];
#pragma unroll
    for (int r = 0; r < 4; ++r) {
      p0[r] = packbf(s0[2 * r], s0[2 * r + 1]);   // own tile0: keys 8h + 2r,2r+1
      p1[r] = packbf(s1[2 * r], s1[2 * r + 1]);   // own tile1: keys 16+8h+2r,2r+1
    }
    V16 bp;
#pragma unroll
    for (int r = 0; r < 4; ++r) {
      unsigned t   = h ? p0[r] : p1[r];           // expose what partner needs
      unsigned rec = __shfl_xor(t, 16, 32);
      bp.w[r]     = h ? rec   : p0[r];            // keys 16h + 0..7
      bp.w[4 + r] = h ? p1[r] : rec;              // keys 16h + 8..15
    }

    // --- O^T += V^T x P^T  (A rows = channels, contiguous from vws)
#pragma unroll
    for (int d = 0; d < 6; ++d) {
      const unsigned short* vrow =
          vws + ((size_t)b * DP + d * 16 + lcol) * Nc + kb;
      V16 av;
      av.u[0] = *(const uint4*)(vrow + koff);
      av.u[1] = *(const uint4*)(vrow + 8 + koff);
      oacc[d] = __builtin_amdgcn_wmma_f32_16x16x32_bf16(false, av.v, false, bp.v,
                                                        (short)0, oacc[d], false, false);
    }
  }

  // --- denominator sits in the ones-channel (72): chunk 4, slot 0, lanes 16-31
  float lv  = oacc[4][0];
  float lvp = __shfl_xor(lv, 16, 32);
  float inv = 1.0f / (h ? lv : lvp);

  // --- D layout: slot r -> channel d*16 + 8h + r (contiguous!): v8f stores
#pragma unroll
  for (int d = 0; d < 6; ++d) {
    v8f outv = oacc[d] * inv;
    float* op = ows + ((size_t)b * Nc + qbase + lcol) * DP + d * 16 + 8 * h;
    *(v8f*)op = outv;
  }
}

// -------------------------------------------------------------------------
// Stage 3: pooled[b,g,n] = sum_f x[b, 4g+f, n] * O[b, n, 4g+f]
// -------------------------------------------------------------------------
__global__ __launch_bounds__(256) void pool_kernel(
    const float* __restrict__ x, const float* __restrict__ ows,
    float* __restrict__ out)
{
  size_t id = (size_t)blockIdx.x * 256 + threadIdx.x;
  if (id >= (size_t)Bc * COo * Nc) return;
  int n = (int)(id % Nc);
  size_t t = id / Nc;
  int g = (int)(t % COo);
  int b = (int)(t / COo);
  const float* op = ows + ((size_t)b * Nc + n) * DP + g * FACT;
  float acc = 0.f;
#pragma unroll
  for (int f = 0; f < FACT; ++f)
    acc = fmaf(x[((size_t)b * Cc + g * FACT + f) * Nc + n], op[f], acc);
  out[id] = acc;
}

// -------------------------------------------------------------------------
extern "C" void kernel_launch(void* const* d_in, const int* in_sizes, int n_in,
                              void* d_out, int out_size, void* d_ws, size_t ws_size,
                              hipStream_t stream)
{
  const float* x  = (const float*)d_in[0];
  const float* Wq = (const float*)d_in[1];
  const float* bq = (const float*)d_in[2];
  const float* Wk = (const float*)d_in[3];
  const float* bk = (const float*)d_in[4];
  const float* Wv = (const float*)d_in[5];
  const float* bv = (const float*)d_in[6];
  float* out = (float*)d_out;

  const size_t nbf = (size_t)Bc * Nc * DP;                  // 1,769,472 elems
  size_t need = 3 * nbf * sizeof(unsigned short) + nbf * sizeof(float);
  if (ws_size < need) return;                               // ~17.7 MB scratch

  unsigned short* qws = (unsigned short*)d_ws;
  unsigned short* kws = qws + nbf;
  unsigned short* vws = kws + nbf;
  float*          ows = (float*)(vws + nbf);

  qkv_kernel<<<(Bc * DP * Nc) / 256, 256, 0, stream>>>(x, Wq, bq, Wk, bk, Wv, bv,
                                                       qws, kws, vws);
  attn_kernel<<<Bc * (Nc / 64), 128, 0, stream>>>(qws, kws, vws, ows);
  pool_kernel<<<((size_t)Bc * COo * Nc + 255) / 256, 256, 0, stream>>>(x, ows, out);
}
